// MultiHeadAttention_65481071408989
// MI455X (gfx1250) — compile-verified, matrix operand order checked by reading, not run on hardware
//
#include <hip/hip_runtime.h>
#include <cstdint>
#include <cstddef>

#define EMBED 1024
#define SEQ   2048
#define BATCH 4
#define HEADS 16
#define DKH   64
#define MTOT  (BATCH * SEQ)   // 8192 token rows

typedef __bf16 bf16_t;
typedef bf16_t v16bf __attribute__((ext_vector_type(16)));
typedef float  v8f   __attribute__((ext_vector_type(8)));
typedef unsigned int   u32;
typedef unsigned short u16;
typedef u32 u32x4 __attribute__((ext_vector_type(4)));
typedef int i32x8 __attribute__((ext_vector_type(8)));
typedef int i32x4 __attribute__((ext_vector_type(4)));
typedef int v4i   __attribute__((ext_vector_type(4)));

typedef __attribute__((address_space(1))) v4i  as1_v4i;
typedef __attribute__((address_space(3))) v4i  as3_v4i;
typedef __attribute__((address_space(3))) void las_void;

union FragU { v16bf v; uint4 q[2]; };

#if defined(__has_builtin)
#if __has_builtin(__builtin_amdgcn_global_load_async_to_lds_b128)
#define HAVE_ASYNC_LDS 1
#endif
#if __has_builtin(__builtin_amdgcn_tensor_load_to_lds)
#define HAVE_TDM 1
#endif
#endif

// LDS byte offset of a generic pointer known to point into shared memory
__device__ __forceinline__ u32 lds_off(const void* p) {
  return (u32)(__SIZE_TYPE__)(las_void*)p;
}

// Async global->LDS 16B copy (per-lane addresses), tracked by ASYNCcnt
__device__ __forceinline__ void async_copy_b128(const bf16_t* g, u16* l) {
#if defined(HAVE_ASYNC_LDS)
  __builtin_amdgcn_global_load_async_to_lds_b128((as1_v4i*)g, (as3_v4i*)l, 0, 0);
#else
  asm volatile("global_load_async_to_lds_b128 %0, %1, off"
               :: "v"(lds_off(l)), "v"((void*)g)
               : "memory");
#endif
}

__device__ __forceinline__ void wait_async0() {
#if defined(__has_builtin) && __has_builtin(__builtin_amdgcn_s_wait_asynccnt)
  __builtin_amdgcn_s_wait_asynccnt(0);
#else
  asm volatile("s_wait_asynccnt 0x0" ::: "memory");
#endif
}

// fp32 -> bf16 round-to-nearest-even (bit trick; NaN edge ignored)
__device__ __forceinline__ u16 f2bf(float f) {
  u32 u = __float_as_uint(f);
  u32 r = u + 0x7FFFu + ((u >> 16) & 1u);
  return (u16)(r >> 16);
}
__device__ __forceinline__ u32 pack2(float lo, float hi) {
  return (u32)f2bf(lo) | ((u32)f2bf(hi) << 16);
}

// ---------------------------------------------------------------------------
// 16x32 bf16 fragment load (A and B operands; K-dimension contiguous).
// Lanes 0-15 -> rows, hold K = {0..7, 16..23}; lanes 16-31 -> same rows,
// K = {8..15, 24..31}. Two 16-byte chunks per lane.
// ---------------------------------------------------------------------------
__device__ __forceinline__ v16bf load_frag_g(const bf16_t* __restrict__ base, int ld) {
  const int lane = threadIdx.x & 31;
  const int r    = lane & 15;
  const int koff = (lane >> 4) << 3;
  const bf16_t* p = base + (size_t)r * ld + koff;
  FragU f;
  f.q[0] = *(const uint4*)(p);
  f.q[1] = *(const uint4*)(p + 16);
  return f.v;
}

__device__ __forceinline__ v16bf load_frag_lds(const u16* base, int ld) {
  const int lane = threadIdx.x & 31;
  const int r    = lane & 15;
  const int koff = (lane >> 4) << 3;
  const u16* p = base + r * ld + koff;
  FragU f;
  f.q[0] = *(const uint4*)(p);
  f.q[1] = *(const uint4*)(p + 16);
  return f.v;
}

__device__ __forceinline__ v8f wmma_bf16(v16bf a, v16bf b, v8f c) {
  return __builtin_amdgcn_wmma_f32_16x16x32_bf16(false, a, false, b, (short)0, c,
                                                 false, false);
}

// ---------------------------------------------------------------------------
// TDM: 2-D tile load, global -> LDS. Descriptor built per the CDNA5 D# spec.
//   data_size = 2 bytes (bf16); tile [tile_d1][tile_d0], dim0 contiguous in
//   memory with row stride stride0 (elements); packed contiguously in LDS.
// ---------------------------------------------------------------------------
#if defined(HAVE_TDM)
__device__ __forceinline__ void tdm_load_2d(u32 ldsByteOff, const void* gaddr,
                                            u32 tensor_d0, u32 tensor_d1,
                                            u32 tile_d0, u32 tile_d1, u32 stride0) {
  const unsigned long long ga = (unsigned long long)(__SIZE_TYPE__)gaddr;
  u32x4 g0;
  g0[0] = 0x1u;                                          // count=1 valid descriptor
  g0[1] = ldsByteOff;                                    // lds_addr
  g0[2] = (u32)ga;                                       // global_addr[31:0]
  g0[3] = (u32)((ga >> 32) & 0x1FFFFFFu) | (2u << 30);   // global_addr[56:32], type=2
  i32x8 g1;
  g1[0] = (int)(1u << 16);                               // data_size=1 (2 bytes)
  g1[1] = (int)((tensor_d0 & 0xFFFFu) << 16);            // tensor_dim0[15:0]
  g1[2] = (int)(((tensor_d0 >> 16) & 0xFFFFu) |
                ((tensor_d1 & 0xFFFFu) << 16));          // dim0[31:16] | dim1[15:0]
  g1[3] = (int)(((tensor_d1 >> 16) & 0xFFFFu) |
                ((tile_d0 & 0xFFFFu) << 16));            // dim1[31:16] | tile_dim0
  g1[4] = (int)(tile_d1 & 0xFFFFu);                      // tile_dim1, tile_dim2=0
  g1[5] = (int)stride0;                                  // tensor_dim0_stride[31:0]
  g1[6] = 0;                                             // stride0[47:32], stride1 lo
  g1[7] = 0;
  const i32x4 z4 = {0, 0, 0, 0};
  const i32x8 z8 = {0, 0, 0, 0, 0, 0, 0, 0};
  __builtin_amdgcn_tensor_load_to_lds(g0, g1, z4, z4, z8, 0);
}
#endif

// ---------------------------------------------------------------------------
// fp32 -> bf16 conversion kernel (8 elements / thread, 16B stores)
// ---------------------------------------------------------------------------
__global__ __launch_bounds__(256)
void cvt_f32_to_bf16(const float* __restrict__ in, u16* __restrict__ out, int n8) {
  int i = blockIdx.x * blockDim.x + threadIdx.x;
  if (i >= n8) return;
  const float4* p = (const float4*)in + (size_t)i * 2;
  float4 a = p[0];
  float4 b = p[1];
  uint4 o;
  o.x = pack2(a.x, a.y);
  o.y = pack2(a.z, a.w);
  o.z = pack2(b.x, b.y);
  o.w = pack2(b.z, b.w);
  ((uint4*)out)[i] = o;
}

// ---------------------------------------------------------------------------
// bf16 WMMA GEMM: C[M,N] = A[M,K] * Bw[N,K]^T  (M=8192, N=K=1024)
// Block = 128 threads = 4 waves (2x2); wave tile 32(M) x 64(N).
// Block tile A 64x32 / B 128x32 double-buffered in LDS, filled with
// global_load_async_to_lds_b128 (ASYNCcnt) and consumed via ds_load_b128.
// MODE 0: fp32 -> outF[M,N];  MODE 1: bf16 -> [B,H,S,Dk];  MODE 2: bf16 -> [B,H,Dk,S]
// ---------------------------------------------------------------------------
#define GEMM_ABUF (64 * 32)
#define GEMM_BBUF (128 * 32)
#define GEMM_BUF  (GEMM_ABUF + GEMM_BBUF)

template<int MODE>
__global__ __launch_bounds__(128)
void gemm_bf16(const bf16_t* __restrict__ A, const bf16_t* __restrict__ Bw,
               float* __restrict__ outF, bf16_t* __restrict__ outB) {
  __shared__ u16 lds[2 * GEMM_BUF];   // 24 KB

  const int tid   = threadIdx.x;
  const int wave  = tid >> 5;
  const int waveM = wave >> 1;
  const int waveN = wave & 1;
  const int mBlk = blockIdx.y * 64;
  const int nBlk = blockIdx.x * 128;
  const int lane = tid & 31;
  const int nl = lane & 15;
  const int mh = (lane >> 4) << 3;

  const v8f vzero = {0.f, 0.f, 0.f, 0.f, 0.f, 0.f, 0.f, 0.f};
  v8f acc[2][4];
#pragma unroll
  for (int i = 0; i < 2; ++i)
#pragma unroll
    for (int j = 0; j < 4; ++j) acc[i][j] = vzero;

  // Issue async copies of the k-step tile into LDS buffer `buf`.
  auto issue_copy = [&](int buf, int k) {
    u16* bA = lds + buf * GEMM_BUF;
    u16* bB = bA + GEMM_ABUF;
    // A: 64 rows x 32 elems = 256 16B chunks; 2 per thread
#pragma unroll
    for (int c = tid; c < 256; c += 128) {
      const int row = c >> 2, col = (c & 3) * 8;
      async_copy_b128(A + (size_t)(mBlk + row) * EMBED + k + col,
                      bA + row * 32 + col);
    }
    // B: 128 rows x 32 elems = 512 16B chunks; 4 per thread
#pragma unroll
    for (int c = tid; c < 512; c += 128) {
      const int row = c >> 2, col = (c & 3) * 8;
      async_copy_b128(Bw + (size_t)(nBlk + row) * EMBED + k + col,
                      bB + row * 32 + col);
    }
  };

  issue_copy(0, 0);
  wait_async0();
  __syncthreads();

  for (int kt = 0; kt < EMBED / 32; ++kt) {
    const int cur = kt & 1;
    if (kt + 1 < EMBED / 32) issue_copy(1 - cur, (kt + 1) * 32);

    const u16* bA = lds + cur * GEMM_BUF;
    const u16* bB = bA + GEMM_ABUF;

    v16bf a0 = load_frag_lds(bA + (waveM * 32 + 0) * 32, 32);
    v16bf a1 = load_frag_lds(bA + (waveM * 32 + 16) * 32, 32);
#pragma unroll
    for (int j = 0; j < 4; ++j) {
      v16bf bfr = load_frag_lds(bB + (waveN * 64 + j * 16) * 32, 32);
      acc[0][j] = wmma_bf16(a0, bfr, acc[0][j]);
      acc[1][j] = wmma_bf16(a1, bfr, acc[1][j]);
    }

    if (kt + 1 < EMBED / 32) {
      wait_async0();      // next tile landed in LDS
      __syncthreads();    // ...for every wave
    }
  }

#pragma unroll
  for (int i = 0; i < 2; ++i)
#pragma unroll
    for (int j = 0; j < 4; ++j)
#pragma unroll
      for (int r = 0; r < 8; ++r) {
        const int m = mBlk + waveM * 32 + i * 16 + mh + r;
        const int n = nBlk + waveN * 64 + j * 16 + nl;
        const float val = acc[i][j][r];
        if (MODE == 0) {
          outF[(size_t)m * EMBED + n] = val;
        } else {
          const int b  = m >> 11;
          const int s  = m & (SEQ - 1);
          const int h  = n >> 6;
          const int dk = n & (DKH - 1);
          size_t idx;
          if (MODE == 1)
            idx = (((size_t)(b * HEADS + h)) * SEQ + s) * DKH + dk;
          else
            idx = (((size_t)(b * HEADS + h)) * DKH + dk) * SEQ + s;
          ((u16*)outB)[idx] = f2bf(val);
        }
      }
}

// ---------------------------------------------------------------------------
// Flash attention: grid = (S/64, B*H), block = 128 threads (4 waves).
// Each wave: 16 query rows, online softmax over 64-key tiles.
// K tile (64 keys x 64 dk) and V^T tile (64 dk x 64 s) staged in LDS by the
// Tensor Data Mover (TENSORcnt) each iteration; P restaged via LDS for the
// P*V WMMA. V is pre-transposed [B,H,Dk,S].
// ---------------------------------------------------------------------------
__global__ __launch_bounds__(128)
void attn_kernel(const bf16_t* __restrict__ qb, const bf16_t* __restrict__ kb,
                 const bf16_t* __restrict__ vtb, bf16_t* __restrict__ aob) {
  __shared__ u16 kTile[64 * 64];     // [key][dk]
  __shared__ u16 vTile[64 * 64];     // [dk][s]
  __shared__ u16 psh[4 * 16 * 72];   // per-wave P tile, row stride 72 (144B)

  const int tid  = threadIdx.x;
  const int wave = tid >> 5;
  const int lane = tid & 31;
  const int nl = lane & 15;
  const int mh = (lane >> 4) << 3;
  const int bh = blockIdx.y;
  const int b = bh >> 4;
  const int h = bh & 15;
  const int q0 = blockIdx.x * 64 + wave * 16;

  const size_t headBase = (size_t)bh * SEQ * DKH;    // K/Q per-head base
  const size_t vHead    = (size_t)bh * DKH * SEQ;    // V^T per-head base

  v16bf qf[2];
#pragma unroll
  for (int ks = 0; ks < 2; ++ks)
    qf[ks] = load_frag_g(qb + headBase + (size_t)q0 * DKH + ks * 32, DKH);

  const v8f vzero = {0.f, 0.f, 0.f, 0.f, 0.f, 0.f, 0.f, 0.f};
  v8f o[4];
#pragma unroll
  for (int j = 0; j < 4; ++j) o[j] = vzero;

  float mrow[8], lrow[8];
#pragma unroll
  for (int r = 0; r < 8; ++r) { mrow[r] = -1e30f; lrow[r] = 0.f; }

  u16* myP = psh + wave * (16 * 72);

  for (int kt = 0; kt < SEQ / 64; ++kt) {
#if defined(HAVE_TDM)
    if (wave == 0) {
      // K tile: rows kt*64..+63 of per-head [SEQ][DKH], dim0 = dk contiguous
      tdm_load_2d(lds_off(kTile), kb + headBase + (size_t)(kt * 64) * DKH,
                  DKH, SEQ, 64, 64, DKH);
      // V^T tile: 64 dk rows x 64 s cols of per-head [DKH][SEQ]
      tdm_load_2d(lds_off(vTile), vtb + vHead + kt * 64,
                  SEQ, DKH, 64, 64, SEQ);
      __builtin_amdgcn_s_wait_tensorcnt(0);
    }
#else
    {
      const bf16_t* kg = kb + headBase + (size_t)(kt * 64) * DKH; // contiguous 8KB
#pragma unroll
      for (int c = tid; c < 256; c += 128)
        *(uint4*)(kTile + c * 8) = *(const uint4*)(kg + c * 8);
      const bf16_t* vg = vtb + vHead + kt * 64;
#pragma unroll
      for (int c = tid; c < 256; c += 128) {
        const int row = c >> 2, col = (c & 3) * 8;
        *(uint4*)(vTile + row * 64 + col) = *(const uint4*)(vg + (size_t)row * SEQ + col);
      }
    }
#endif
    __syncthreads();   // K/V tiles visible to all waves

    // ---- scores S = Q * K^T ----
    v8f sc[4];
#pragma unroll
    for (int j = 0; j < 4; ++j) sc[j] = vzero;
#pragma unroll
    for (int ks = 0; ks < 2; ++ks) {
#pragma unroll
      for (int j = 0; j < 4; ++j) {
        v16bf kf = load_frag_lds(kTile + (j * 16) * 64 + ks * 32, 64);
        sc[j] = wmma_bf16(qf[ks], kf, sc[j]);
      }
    }
#pragma unroll
    for (int j = 0; j < 4; ++j) sc[j] = sc[j] * 0.125f;   // 1/sqrt(Dk)

    // ---- online softmax ----
    float alpha[8];
#pragma unroll
    for (int r = 0; r < 8; ++r) {
      float t = fmaxf(fmaxf(sc[0][r], sc[1][r]), fmaxf(sc[2][r], sc[3][r]));
#pragma unroll
      for (int off = 1; off < 16; off <<= 1)
        t = fmaxf(t, __shfl_xor(t, off, 32));
      float mn = fmaxf(mrow[r], t);
      alpha[r] = __expf(mrow[r] - mn);
      mrow[r] = mn;
    }

#pragma unroll
    for (int j = 0; j < 4; ++j) {
#pragma unroll
      for (int r = 0; r < 8; ++r) {
        float p = __expf(sc[j][r] - mrow[r]);
        sc[j][r] = p;
        myP[(mh + r) * 72 + j * 16 + nl] = f2bf(p);
      }
    }

#pragma unroll
    for (int r = 0; r < 8; ++r) {
      float s = (sc[0][r] + sc[1][r]) + (sc[2][r] + sc[3][r]);
#pragma unroll
      for (int off = 1; off < 16; off <<= 1)
        s += __shfl_xor(s, off, 32);
      lrow[r] = lrow[r] * alpha[r] + s;
    }
#pragma unroll
    for (int j = 0; j < 4; ++j)
#pragma unroll
      for (int r = 0; r < 8; ++r)
        o[j][r] *= alpha[r];

    __syncthreads();   // P tile visible

    // ---- O += P * V ----
#pragma unroll
    for (int ks = 0; ks < 2; ++ks) {
      v16bf pa = load_frag_lds(myP + ks * 32, 72);
#pragma unroll
      for (int j = 0; j < 4; ++j) {
        v16bf vf = load_frag_lds(vTile + (j * 16) * 64 + ks * 32, 64);
        o[j] = wmma_bf16(pa, vf, o[j]);
      }
    }

    __syncthreads();   // all reads done before next tile overwrites LDS
  }

  // ---- normalize and store bf16 attention output [B,S,E] ----
  float inv[8];
#pragma unroll
  for (int r = 0; r < 8; ++r) inv[r] = 1.0f / lrow[r];
#pragma unroll
  for (int j = 0; j < 4; ++j) {
#pragma unroll
    for (int r = 0; r < 8; ++r) {
      float val = o[j][r] * inv[r];
      const int s = q0 + mh + r;
      const int e = h * DKH + j * 16 + nl;
      ((u16*)aob)[((size_t)b * SEQ + s) * EMBED + e] = f2bf(val);
    }
  }
}

// ---------------------------------------------------------------------------
// Host-side orchestration
// ---------------------------------------------------------------------------
extern "C" void kernel_launch(void* const* d_in, const int* in_sizes, int n_in,
                              void* d_out, int out_size, void* d_ws, size_t ws_size,
                              hipStream_t stream) {
  const float* x  = (const float*)d_in[0];
  const float* wq = (const float*)d_in[1];
  const float* wk = (const float*)d_in[2];
  const float* wv = (const float*)d_in[3];
  const float* wo = (const float*)d_in[4];

  const size_t NX = (size_t)MTOT * EMBED;
  const size_t NW = (size_t)EMBED * EMBED;

  bf16_t* wp  = (bf16_t*)d_ws;
  bf16_t* xb  = wp;  wp += NX;
  bf16_t* wqb = wp;  wp += NW;
  bf16_t* wkb = wp;  wp += NW;
  bf16_t* wvb = wp;  wp += NW;
  bf16_t* wob = wp;  wp += NW;
  bf16_t* qb  = wp;  wp += NX;
  bf16_t* kb  = wp;  wp += NX;
  bf16_t* vtb = wp;  wp += NX;
  bf16_t* aob = wp;  wp += NX;

  {
    int n8 = (int)(NX / 8);
    cvt_f32_to_bf16<<<(n8 + 255) / 256, 256, 0, stream>>>(x, (u16*)xb, n8);
    n8 = (int)(NW / 8);
    dim3 g((n8 + 255) / 256);
    cvt_f32_to_bf16<<<g, 256, 0, stream>>>(wq, (u16*)wqb, n8);
    cvt_f32_to_bf16<<<g, 256, 0, stream>>>(wk, (u16*)wkb, n8);
    cvt_f32_to_bf16<<<g, 256, 0, stream>>>(wv, (u16*)wvb, n8);
    cvt_f32_to_bf16<<<g, 256, 0, stream>>>(wo, (u16*)wob, n8);
  }

  dim3 ggrid(EMBED / 128, MTOT / 64);   // (8, 128)
  gemm_bf16<1><<<ggrid, 128, 0, stream>>>(xb, wqb, nullptr, qb);   // Q [B,H,S,Dk]
  gemm_bf16<1><<<ggrid, 128, 0, stream>>>(xb, wkb, nullptr, kb);   // K [B,H,S,Dk]
  gemm_bf16<2><<<ggrid, 128, 0, stream>>>(xb, wvb, nullptr, vtb);  // V [B,H,Dk,S]

  dim3 agrid(SEQ / 64, BATCH * HEADS);  // (32, 64)
  attn_kernel<<<agrid, 128, 0, stream>>>(qb, kb, vtb, aob);

  gemm_bf16<0><<<ggrid, 128, 0, stream>>>(aob, wob, (float*)d_out, nullptr);
}
